// TemporalFusionModule_19567871001296
// MI455X (gfx1250) — matrix operand order repaired and checked by reading, not compiled, not run on hardware
//
#include <hip/hip_runtime.h>
#include <cstdint>

typedef _Float16 v16h __attribute__((ext_vector_type(16)));
typedef float    v8f  __attribute__((ext_vector_type(8)));

// ---------------------------------------------------------------------------
// frames (4,16,3,256,256) f32 -> (4,3,16,256,256) f16, one elem/thread
__global__ void k_frames_f16(const float* __restrict__ src, _Float16* __restrict__ dst) {
  constexpr int T = 16, C = 3, H = 256, W = 256;
  int i = blockIdx.x * blockDim.x + threadIdx.x;
  if (i >= 4 * C * T * H * W) return;
  int w = i & (W - 1);
  int h = (i >> 8) & (H - 1);
  int t = (i >> 16) & (T - 1);
  int c = (i >> 20) % C;
  int b = i / (C * T * H * W);
  dst[i] = (_Float16)src[(((size_t)(b * T + t) * C + c) * H + h) * W + w];
}

// conv weight (Cout, Cin*27) f32 -> K-padded f16
__global__ void k_wpad_f16(const float* __restrict__ w, _Float16* __restrict__ wp,
                           int Cout, int Ktot, int Kpad) {
  int i = blockIdx.x * blockDim.x + threadIdx.x;
  if (i >= Cout * Kpad) return;
  int k = i % Kpad, co = i / Kpad;
  wp[i] = (k < Ktot) ? (_Float16)w[co * Ktot + k] : (_Float16)0.0f;
}

__global__ void k_f32_to_f16(const float* __restrict__ s, _Float16* __restrict__ d, int n) {
  int i = blockIdx.x * blockDim.x + threadIdx.x;
  if (i < n) d[i] = (_Float16)s[i];
}

// ---------------------------------------------------------------------------
// Implicit-GEMM 3x3x3 conv. One wave owns a 16(M = w-positions) x COUT tile:
// NT = COUT/16 accumulators, A fragment loaded once per K-step and reused
// across NT WMMAs. B fragments for all NT tiles are loaded as a batch first
// so the WMMA chain issues back-to-back while later loads are still in flight.
template <int CIN, int COUT, int HW_, int NT>
__global__ void __launch_bounds__(32) k_conv_wmma(const _Float16* __restrict__ in,
                                                  const _Float16* __restrict__ wp,
                                                  const float* __restrict__ bias,
                                                  _Float16* __restrict__ out) {
  constexpr int T = 16, H = HW_, W = HW_;
  constexpr int Ktot = CIN * 27;
  constexpr int Kpad = (Ktot + 31) & ~31;
  constexpr int Wt = W / 16;
  int lane = threadIdx.x & 31;
  int m = lane & 15, lh = lane >> 4;
  int bx = blockIdx.x;
  int wt = bx % Wt; bx /= Wt;
  int h = bx % H;  bx /= H;
  int t = bx % T;
  int b = bx / T;
  int w0 = wt << 4;

  v8f zero = {};
  v8f acc[NT];
#pragma unroll
  for (int n = 0; n < NT; ++n) acc[n] = zero;

  for (int kb = 0; kb < Kpad; kb += 32) {
    // batch all B fragments first (independent b128 loads)
    v16h bf[NT];
#pragma unroll
    for (int n = 0; n < NT; ++n)
      bf[n] = *(const v16h*)(wp + (size_t)(n * 16 + m) * Kpad + kb + lh * 16);

    v16h a;
#pragma unroll
    for (int e = 0; e < 16; ++e) {
      int v = e >> 1, pos = e & 1;
      int kl = lh * 16 + 2 * v + pos;
      int k = kb + kl;
      float val = 0.f;
      if (Ktot == Kpad || k < Ktot) {   // folds away when K is a multiple of 32
        int ci = k / 27, r = k % 27;    // constant divisors -> mul/shift
        int kt = r / 9; r %= 9;
        int kh = r / 3, kw = r % 3;
        int tt = t + kt - 1, hh = h + kh - 1, ww = w0 + m + kw - 1;
        if ((unsigned)tt < (unsigned)T && (unsigned)hh < (unsigned)H &&
            (unsigned)ww < (unsigned)W)
          val = (float)in[(((size_t)(b * CIN + ci) * T + tt) * H + hh) * (size_t)W + ww];
      }
      a[e] = (_Float16)val;
    }
#pragma unroll
    for (int n = 0; n < NT; ++n)
      acc[n] = __builtin_amdgcn_wmma_f32_16x16x32_f16(false, a, false, bf[n], (short)0,
                                                      acc[n], false, false);
  }
#pragma unroll
  for (int n = 0; n < NT; ++n) {
    int co = n * 16 + m;
    float bv = bias[co];
    size_t obase = (((size_t)(b * COUT + co) * T + t) * H + h) * (size_t)W + w0;
#pragma unroll
    for (int v = 0; v < 8; ++v) out[obase + v + lh * 8] = (_Float16)(acc[n][v] + bv);
  }
}

// per-channel batch stats over (B, T*H*W)
__global__ void k_bn_stats(const _Float16* __restrict__ raw, float* __restrict__ stats,
                           int B, int Cout, long THW) {
  int c = blockIdx.x;
  __shared__ float ssum[256], ssq[256];
  float s = 0.f, q = 0.f;
  for (int b = 0; b < B; ++b) {
    const _Float16* p = raw + ((size_t)b * Cout + c) * THW;
    for (long i = threadIdx.x; i < THW; i += blockDim.x) {
      float v = (float)p[i];
      s += v; q += v * v;
    }
  }
  ssum[threadIdx.x] = s; ssq[threadIdx.x] = q;
  __syncthreads();
  for (int st = 128; st > 0; st >>= 1) {
    if ((int)threadIdx.x < st) {
      ssum[threadIdx.x] += ssum[threadIdx.x + st];
      ssq[threadIdx.x]  += ssq[threadIdx.x + st];
    }
    __syncthreads();
  }
  if (threadIdx.x == 0) {
    float n = (float)B * (float)THW;
    float mean = ssum[0] / n;
    float var = ssq[0] / n - mean * mean;
    stats[c] = mean;
    stats[Cout + c] = rsqrtf(var + 1e-5f);
  }
}

// BN + ReLU + 2x2 spatial maxpool, f16 out, one output elem/thread
template <int COUT, int H, int W>
__global__ void k_bn_pool(const _Float16* __restrict__ raw, _Float16* __restrict__ out,
                          const float* __restrict__ gamma, const float* __restrict__ beta,
                          const float* __restrict__ stats) {
  constexpr int T = 16, Ho = H / 2, Wo = W / 2;
  int i = blockIdx.x * blockDim.x + threadIdx.x;
  if (i >= 4 * COUT * T * Ho * Wo) return;
  int wo = i % Wo;
  int ho = (i / Wo) % Ho;
  int t = (i / (Wo * Ho)) % T;
  int c = (i / (Wo * Ho * T)) % COUT;
  int b = i / (Wo * Ho * T * COUT);
  float mean = stats[c], inv = stats[COUT + c], g = gamma[c], be = beta[c];
  size_t base = (((size_t)(b * COUT + c) * T + t) * H + 2 * ho) * (size_t)W + 2 * wo;
  float mx = 0.f;  // relu(.) >= 0 so max of 4 relu values starts at 0
#pragma unroll
  for (int dy = 0; dy < 2; ++dy)
#pragma unroll
    for (int dx = 0; dx < 2; ++dx) {
      float v = (float)raw[base + (size_t)dy * W + dx];
      v = (v - mean) * inv * g + be;
      v = v > 0.f ? v : 0.f;
      mx = v > mx ? v : mx;
    }
  out[i] = (_Float16)mx;
}

// ---------------------------------------------------------------------------
// bbox MLP -> U = bbox_feat + enc_b  (enc GEMM atomically adds frame_feat)
__global__ void k_bbox_u(const float* __restrict__ tb,
                         const float* __restrict__ w1, const float* __restrict__ b1,
                         const float* __restrict__ w2, const float* __restrict__ b2,
                         const float* __restrict__ encb, float* __restrict__ U) {
  constexpr int T = 16, D = 512;
  __shared__ float hid[64];
  int r = blockIdx.x;
  int b = r / T, t = r % T;
  int ts = (t == 0) ? 0 : (t - 1);
  if (threadIdx.x < 64) {
    float a = b1[threadIdx.x];
#pragma unroll
    for (int i = 0; i < 4; ++i) a += tb[((size_t)b * T + ts) * 4 + i] * w1[threadIdx.x * 4 + i];
    hid[threadIdx.x] = a > 0.f ? a : 0.f;
  }
  __syncthreads();
  int o = threadIdx.x;
  float a = b2[o] + encb[o];
  for (int j = 0; j < 64; ++j) a += hid[j] * w2[o * 64 + j];
  U[(size_t)r * D + o] = a;
}

// enc_proj GEMM: xf(64 x 131072) @ W^T(131072 x 512).
// Split-K (64 chunks) x 8 groups of 4 N-tiles; atomic f32 accumulation.
__global__ void __launch_bounds__(32) k_enc_gemm(const _Float16* __restrict__ xf,
                                                 const _Float16* __restrict__ wq,
                                                 float* __restrict__ U) {
  constexpr int FLAT = 131072, NT = 4, KCH = 2048;
  int lane = threadIdx.x & 31;
  int m = lane & 15, lh = lane >> 4;
  int mt = blockIdx.x, ng = blockIdx.y;
  int kb0 = blockIdx.z * KCH;
  int rrow = mt * 16 + m;
  int b = rrow >> 4, t = rrow & 15;
  // xf element (b,c,t,hw) = xf[b*2097152 + c*16384 + t*1024 + hw]
  const _Float16* abase = xf + (size_t)b * 2097152 + (size_t)t * 1024;

  v8f zero = {};
  v8f acc[NT];
#pragma unroll
  for (int n = 0; n < NT; ++n) acc[n] = zero;

  for (int kb = kb0; kb < kb0 + KCH; kb += 32) {
    // batch all B fragments (independent b128 loads) + stream prefetch
    v16h bf[NT];
#pragma unroll
    for (int n = 0; n < NT; ++n) {
      const _Float16* wrow = wq + (size_t)((ng * NT + n) * 16 + m) * FLAT;
      __builtin_prefetch(wrow + kb + 2048, 0, 0);  // global_prefetch_b8
      bf[n] = *(const v16h*)(wrow + kb + lh * 16);
    }
    v16h a;
#pragma unroll
    for (int e = 0; e < 16; ++e) {
      int v = e >> 1, pos = e & 1;
      int kl = lh * 16 + 2 * v + pos;
      int k = kb + kl;
      int c = k >> 10, hw = k & 1023;
      a[e] = abase[(size_t)c * 16384 + hw];
    }
#pragma unroll
    for (int n = 0; n < NT; ++n)
      acc[n] = __builtin_amdgcn_wmma_f32_16x16x32_f16(false, a, false, bf[n], (short)0,
                                                      acc[n], false, false);
  }
#pragma unroll
  for (int n = 0; n < NT; ++n) {
    int co = (ng * NT + n) * 16 + m;
#pragma unroll
    for (int v = 0; v < 8; ++v) {
      int rr = mt * 16 + v + lh * 8;
      atomicAdd(&U[(size_t)rr * 512 + co], acc[n][v]);
    }
  }
}

// ---------------------------------------------------------------------------
// mamba small kernels (compute negligible: plain VALU)
__global__ void k_in_proj(const float* __restrict__ U, const float* __restrict__ w,
                          float* __restrict__ XZ) {
  int i = blockIdx.x * blockDim.x + threadIdx.x;
  if (i >= 64 * 2048) return;
  int r = i >> 11, j = i & 2047;
  const float* u = U + (size_t)r * 512;
  const float* ww = w + (size_t)j * 512;
  float a = 0.f;
  for (int k = 0; k < 512; ++k) a += u[k] * ww[k];
  XZ[i] = a;
}

__global__ void k_mamba_conv(const float* __restrict__ XZ, const float* __restrict__ cw,
                             const float* __restrict__ cb, float* __restrict__ XC) {
  int i = blockIdx.x * blockDim.x + threadIdx.x;
  if (i >= 64 * 1024) return;
  int r = i >> 10, d = i & 1023;
  int b = r >> 4, t = r & 15;
  float a = cb[d];
#pragma unroll
  for (int k = 0; k < 4; ++k) {
    int tt = t + k - 3;
    if (tt >= 0) a += cw[d * 4 + k] * XZ[((size_t)(b * 16 + tt)) * 2048 + d];
  }
  XC[i] = a / (1.f + __expf(-a));
}

__global__ void k_x_proj(const float* __restrict__ XC, const float* __restrict__ w,
                         float* __restrict__ XD) {
  int i = blockIdx.x * blockDim.x + threadIdx.x;
  if (i >= 64 * 64) return;
  int r = i >> 6, j = i & 63;
  float a = 0.f;
  for (int d = 0; d < 1024; ++d) a += XC[(size_t)r * 1024 + d] * w[(size_t)j * 1024 + d];
  XD[i] = a;
}

__global__ void k_dt(const float* __restrict__ XD, const float* __restrict__ w,
                     const float* __restrict__ bia, float* __restrict__ DT) {
  int i = blockIdx.x * blockDim.x + threadIdx.x;
  if (i >= 64 * 1024) return;
  int r = i >> 10, d = i & 1023;
  float a = bia[d];
  for (int q = 0; q < 32; ++q) a += XD[r * 64 + q] * w[d * 32 + q];
  DT[i] = (a > 20.f) ? a : log1pf(__expf(a));
}

__global__ void k_scan(const float* __restrict__ DT, const float* __restrict__ XC,
                       const float* __restrict__ XD, const float* __restrict__ XZ,
                       const float* __restrict__ Alog, const float* __restrict__ Dp,
                       float* __restrict__ Y) {
  int i = blockIdx.x * blockDim.x + threadIdx.x;
  if (i >= 4 * 1024) return;
  int b = i >> 10, d = i & 1023;
  float A[16], h[16];
#pragma unroll
  for (int n = 0; n < 16; ++n) { A[n] = -__expf(Alog[d * 16 + n]); h[n] = 0.f; }
  float Dd = Dp[d];
  for (int t = 0; t < 16; ++t) {
    int r = b * 16 + t;
    float dtv = DT[(size_t)r * 1024 + d];
    float xcv = XC[(size_t)r * 1024 + d];
    float zv = XZ[(size_t)r * 2048 + 1024 + d];
    float acc = 0.f;
#pragma unroll
    for (int n = 0; n < 16; ++n) {
      float hn = __expf(dtv * A[n]) * h[n] + dtv * XD[r * 64 + 32 + n] * xcv;
      h[n] = hn;
      acc += hn * XD[r * 64 + 48 + n];
    }
    Y[(size_t)r * 1024 + d] = (acc + Dd * xcv) * (zv / (1.f + __expf(-zv)));
  }
}

__global__ void k_out_proj(const float* __restrict__ Y, const float* __restrict__ w,
                           float* __restrict__ MO) {
  int i = blockIdx.x * blockDim.x + threadIdx.x;
  if (i >= 64 * 512) return;
  int r = i >> 9, dd = i & 511;
  float a = 0.f;
  for (int d = 0; d < 1024; ++d) a += Y[(size_t)r * 1024 + d] * w[(size_t)dd * 1024 + d];
  MO[i] = a;
}

__global__ void k_head(const float* __restrict__ MO, const float* __restrict__ hw,
                       const float* __restrict__ hb, float* __restrict__ meas) {
  int i = threadIdx.x;
  if (i >= 256) return;
  int r = i >> 2, o = i & 3;
  float a = hb[o];
  for (int dd = 0; dd < 512; ++dd) a += MO[(size_t)r * 512 + dd] * hw[o * 512 + dd];
  meas[i] = a;
}

// ---------------------------------------------------------------------------
// Kalman filter: one thread per batch, 16 sequential steps, 4x4 Gauss-Jordan solve
__global__ void k_kalman(const float* __restrict__ tb, const float* __restrict__ meas,
                         const float* __restrict__ Fm, const float* __restrict__ Hm,
                         const float* __restrict__ logq, const float* __restrict__ logr,
                         float* __restrict__ preds) {
  int b = threadIdx.x;
  if (b >= 4) return;
  float F[8][8], Hh[4][8], q[8], r4[4];
  for (int i = 0; i < 8; ++i)
    for (int j = 0; j < 8; ++j) F[i][j] = Fm[i * 8 + j];
  for (int i = 0; i < 4; ++i)
    for (int j = 0; j < 8; ++j) Hh[i][j] = Hm[i * 8 + j];
  for (int i = 0; i < 8; ++i) q[i] = __expf(logq[i]) + 1e-6f;
  for (int i = 0; i < 4; ++i) r4[i] = __expf(logr[i]) + 1e-6f;
  float s[8], P[8][8];
  for (int i = 0; i < 4; ++i) {
    s[i] = tb[(b * 16 + 0) * 4 + i];
    s[4 + i] = tb[(b * 16 + 1) * 4 + i] - tb[(b * 16 + 0) * 4 + i];
  }
  for (int i = 0; i < 8; ++i)
    for (int j = 0; j < 8; ++j) P[i][j] = (i == j) ? 0.01f : 0.f;

  for (int t = 0; t < 16; ++t) {
    float sp[8];
    for (int i = 0; i < 8; ++i) {
      float a = 0.f;
      for (int j = 0; j < 8; ++j) a += F[i][j] * s[j];
      sp[i] = a;
    }
    float tmp[8][8], Pp[8][8];
    for (int i = 0; i < 8; ++i)
      for (int k = 0; k < 8; ++k) {
        float a = 0.f;
        for (int j = 0; j < 8; ++j) a += F[i][j] * P[j][k];
        tmp[i][k] = a;
      }
    for (int i = 0; i < 8; ++i)
      for (int l = 0; l < 8; ++l) {
        float a = 0.f;
        for (int k = 0; k < 8; ++k) a += tmp[i][k] * F[l][k];
        Pp[i][l] = a + ((i == l) ? q[i] : 0.f);
      }
    float yres[4];
    for (int i = 0; i < 4; ++i) {
      float a = 0.f;
      for (int j = 0; j < 8; ++j) a += Hh[i][j] * sp[j];
      yres[i] = meas[(b * 16 + t) * 4 + i] - a;
    }
    float hp[4][8], S[4][4];
    for (int i = 0; i < 4; ++i)
      for (int k = 0; k < 8; ++k) {
        float a = 0.f;
        for (int j = 0; j < 8; ++j) a += Hh[i][j] * Pp[j][k];
        hp[i][k] = a;
      }
    for (int i = 0; i < 4; ++i)
      for (int l = 0; l < 4; ++l) {
        float a = 0.f;
        for (int k = 0; k < 8; ++k) a += hp[i][k] * Hh[l][k];
        S[i][l] = a + ((i == l) ? r4[i] : 0.f);
      }
    float Z[4][8];
    for (int i = 0; i < 4; ++i)
      for (int j = 0; j < 8; ++j) Z[i][j] = Hh[i][j];
    for (int c = 0; c < 4; ++c) {
      int p = c;
      for (int rr = c + 1; rr < 4; ++rr)
        if (fabsf(S[rr][c]) > fabsf(S[p][c])) p = rr;
      if (p != c) {
        for (int j = 0; j < 4; ++j) { float sw = S[c][j]; S[c][j] = S[p][j]; S[p][j] = sw; }
        for (int j = 0; j < 8; ++j) { float sw = Z[c][j]; Z[c][j] = Z[p][j]; Z[p][j] = sw; }
      }
      float inv = 1.f / S[c][c];
      for (int j = 0; j < 4; ++j) S[c][j] *= inv;
      for (int j = 0; j < 8; ++j) Z[c][j] *= inv;
      for (int rr = 0; rr < 4; ++rr)
        if (rr != c) {
          float f = S[rr][c];
          for (int j = 0; j < 4; ++j) S[rr][j] -= f * S[c][j];
          for (int j = 0; j < 8; ++j) Z[rr][j] -= f * Z[c][j];
        }
    }
    float K[8][4];
    for (int i = 0; i < 8; ++i)
      for (int j = 0; j < 4; ++j) {
        float a = 0.f;
        for (int k = 0; k < 8; ++k) a += Pp[i][k] * Z[j][k];
        K[i][j] = a;
      }
    float sn[8];
    for (int i = 0; i < 8; ++i) {
      float a = sp[i];
      for (int j = 0; j < 4; ++j) a += K[i][j] * yres[j];
      sn[i] = a;
    }
    float IKH[8][8];
    for (int i = 0; i < 8; ++i)
      for (int j = 0; j < 8; ++j) {
        float a = (i == j) ? 1.f : 0.f;
        for (int k = 0; k < 4; ++k) a -= K[i][k] * Hh[k][j];
        IKH[i][j] = a;
      }
    float t1[8][8];
    for (int i = 0; i < 8; ++i)
      for (int k = 0; k < 8; ++k) {
        float a = 0.f;
        for (int j = 0; j < 8; ++j) a += IKH[i][j] * Pp[j][k];
        t1[i][k] = a;
      }
    for (int i = 0; i < 8; ++i)
      for (int l = 0; l < 8; ++l) {
        float a = 0.f;
        for (int k = 0; k < 8; ++k) a += t1[i][k] * IKH[l][k];
        for (int j = 0; j < 4; ++j) a += K[i][j] * r4[j] * K[l][j];
        P[i][l] = a;
      }
    for (int i = 0; i < 4; ++i) preds[(b * 16 + t) * 4 + i] = sn[i];
    for (int i = 0; i < 8; ++i) s[i] = sn[i];
  }
}

// ---------------------------------------------------------------------------
extern "C" void kernel_launch(void* const* d_in, const int* in_sizes, int n_in,
                              void* d_out, int out_size, void* d_ws, size_t ws_size,
                              hipStream_t stream) {
  (void)in_sizes; (void)n_in; (void)out_size; (void)ws_size;
  const float* frames = (const float*)d_in[0];
  const float* tb     = (const float*)d_in[1];
  const float* cw[3]  = {(const float*)d_in[2], (const float*)d_in[6], (const float*)d_in[10]};
  const float* cb[3]  = {(const float*)d_in[3], (const float*)d_in[7], (const float*)d_in[11]};
  const float* cg[3]  = {(const float*)d_in[4], (const float*)d_in[8], (const float*)d_in[12]};
  const float* cbe[3] = {(const float*)d_in[5], (const float*)d_in[9], (const float*)d_in[13]};
  const float* enc_w = (const float*)d_in[14];
  const float* enc_b = (const float*)d_in[15];
  const float* bw1 = (const float*)d_in[16];
  const float* bb1 = (const float*)d_in[17];
  const float* bw2 = (const float*)d_in[18];
  const float* bb2 = (const float*)d_in[19];
  const float* m_inw  = (const float*)d_in[20];
  const float* m_cw   = (const float*)d_in[21];
  const float* m_cb   = (const float*)d_in[22];
  const float* m_xw   = (const float*)d_in[23];
  const float* m_dtw  = (const float*)d_in[24];
  const float* m_dtb  = (const float*)d_in[25];
  const float* m_alog = (const float*)d_in[26];
  const float* m_D    = (const float*)d_in[27];
  const float* m_ow   = (const float*)d_in[28];
  const float* h_w  = (const float*)d_in[29];
  const float* h_b  = (const float*)d_in[30];
  const float* kF   = (const float*)d_in[31];
  const float* kH   = (const float*)d_in[32];
  const float* klq  = (const float*)d_in[33];
  const float* klr  = (const float*)d_in[34];
  float* out = (float*)d_out;

  char* ws = (char*)d_ws;
  size_t off = 0;
  auto alloc = [&](size_t bytes) -> char* {
    size_t a = off;
    off += (bytes + 255) & ~(size_t)255;
    return ws + a;
  };
  _Float16* FR   = (_Float16*)alloc((size_t)4 * 3 * 16 * 256 * 256 * 2);
  _Float16* WP   = (_Float16*)alloc((size_t)128 * 1728 * 2);
  _Float16* RAW  = (_Float16*)alloc((size_t)4 * 32 * 16 * 256 * 256 * 2);
  _Float16* PA   = (_Float16*)alloc((size_t)4 * 32 * 16 * 128 * 128 * 2);
  _Float16* PB   = (_Float16*)alloc((size_t)4 * 64 * 16 * 64 * 64 * 2);
  float*    STAT = (float*)alloc(2 * 128 * 4);
  _Float16* EW   = (_Float16*)alloc((size_t)512 * 131072 * 2);
  float* U   = (float*)alloc(64 * 512 * 4);
  float* XZ  = (float*)alloc(64 * 2048 * 4);
  float* XC  = (float*)alloc(64 * 1024 * 4);
  float* XD  = (float*)alloc(64 * 64 * 4);
  float* DTb = (float*)alloc(64 * 1024 * 4);
  float* Yb  = (float*)alloc(64 * 1024 * 4);
  float* MO  = (float*)alloc(64 * 512 * 4);

  // frames -> f16 NCDHW
  k_frames_f16<<<dim3((4 * 3 * 16 * 256 * 256 + 255) / 256), dim3(256), 0, stream>>>(frames, FR);

  // ---- conv layer 1: 3 -> 32, 256x256 ----
  k_wpad_f16<<<dim3((32 * 96 + 255) / 256), dim3(256), 0, stream>>>(cw[0], WP, 32, 81, 96);
  k_conv_wmma<3, 32, 256, 2><<<dim3(4 * 16 * 256 * 16), dim3(32), 0, stream>>>(FR, WP, cb[0], RAW);
  k_bn_stats<<<dim3(32), dim3(256), 0, stream>>>(RAW, STAT, 4, 32, (long)16 * 256 * 256);
  k_bn_pool<32, 256, 256><<<dim3((4 * 32 * 16 * 128 * 128 + 255) / 256), dim3(256), 0, stream>>>(
      RAW, PA, cg[0], cbe[0], STAT);

  // ---- conv layer 2: 32 -> 64, 128x128 ----
  k_wpad_f16<<<dim3((64 * 864 + 255) / 256), dim3(256), 0, stream>>>(cw[1], WP, 64, 864, 864);
  k_conv_wmma<32, 64, 128, 4><<<dim3(4 * 16 * 128 * 8), dim3(32), 0, stream>>>(PA, WP, cb[1], RAW);
  k_bn_stats<<<dim3(64), dim3(256), 0, stream>>>(RAW, STAT, 4, 64, (long)16 * 128 * 128);
  k_bn_pool<64, 128, 128><<<dim3((4 * 64 * 16 * 64 * 64 + 255) / 256), dim3(256), 0, stream>>>(
      RAW, PB, cg[1], cbe[1], STAT);

  // ---- conv layer 3: 64 -> 128, 64x64 ----
  k_wpad_f16<<<dim3((128 * 1728 + 255) / 256), dim3(256), 0, stream>>>(cw[2], WP, 128, 1728, 1728);
  k_conv_wmma<64, 128, 64, 8><<<dim3(4 * 16 * 64 * 4), dim3(32), 0, stream>>>(PB, WP, cb[2], RAW);
  k_bn_stats<<<dim3(128), dim3(256), 0, stream>>>(RAW, STAT, 4, 128, (long)16 * 64 * 64);
  k_bn_pool<128, 64, 64><<<dim3((4 * 128 * 16 * 32 * 32 + 255) / 256), dim3(256), 0, stream>>>(
      RAW, PA, cg[2], cbe[2], STAT);

  // U = bbox_feat + enc_b
  k_bbox_u<<<dim3(64), dim3(512), 0, stream>>>(tb, bw1, bb1, bw2, bb2, enc_b, U);
  // enc weights -> f16, then split-K WMMA GEMM accumulating into U
  k_f32_to_f16<<<dim3((512 * 131072 + 255) / 256), dim3(256), 0, stream>>>(enc_w, EW,
                                                                           512 * 131072);
  k_enc_gemm<<<dim3(4, 8, 64), dim3(32), 0, stream>>>(PA, EW, U);

  // mamba
  k_in_proj<<<dim3((64 * 2048 + 255) / 256), dim3(256), 0, stream>>>(U, m_inw, XZ);
  k_mamba_conv<<<dim3((64 * 1024 + 255) / 256), dim3(256), 0, stream>>>(XZ, m_cw, m_cb, XC);
  k_x_proj<<<dim3((64 * 64 + 255) / 256), dim3(256), 0, stream>>>(XC, m_xw, XD);
  k_dt<<<dim3((64 * 1024 + 255) / 256), dim3(256), 0, stream>>>(XD, m_dtw, m_dtb, DTb);
  k_scan<<<dim3((4096 + 255) / 256), dim3(256), 0, stream>>>(DTb, XC, XD, XZ, m_alog, m_D, Yb);
  k_out_proj<<<dim3((64 * 512 + 255) / 256), dim3(256), 0, stream>>>(Yb, m_ow, MO);
  k_head<<<dim3(1), dim3(256), 0, stream>>>(MO, h_w, h_b, out + 256);

  // Kalman: preds -> out[0:256], reads meas from out[256:512]
  k_kalman<<<dim3(1), dim3(32), 0, stream>>>(tb, out + 256, kF, kH, klq, klr, out);
}
